// GatedAttentionLayer_28776280883835
// MI455X (gfx1250) — matrix operand order repaired and checked by reading, not compiled, not run on hardware
//
#include <hip/hip_runtime.h>
#include <hip/hip_bf16.h>
#include <math.h>

typedef __bf16 bf16;
typedef __attribute__((ext_vector_type(16))) __bf16 bfx16;
typedef __attribute__((ext_vector_type(8)))  __bf16 bfx8;
typedef __attribute__((ext_vector_type(8)))  float  fx8;

#define BM 256           // M tile per workgroup
#define BN 128           // N tile per workgroup
#define BK 32            // K step (one WMMA K)
#define LDSS 40          // padded LDS row stride in bf16 elements (80B = 5x16B)
#define SEQ 2048
#define DIM 1024
#define NB  8

// epilogue selectors
#define EPI_BF16_BIAS      0
#define EPI_BF16_BIAS_SIG  1
#define EPI_BF16_BIAS_VT   2
#define EPI_F32_RAW        3
#define EPI_BF16_GATE      4
#define EPI_F32_BIAS       5

// ---------------------------------------------------------------------------
// CDNA5 async global->LDS copy (ASYNCcnt-tracked). INST_OFFSET adds to BOTH
// the LDS dest VGPR and the global src address (ISA 10.x async addressing),
// so one base pair covers a 64B row with offset:{0,16,32,48}.
// ---------------------------------------------------------------------------
__device__ __forceinline__ uint32_t lds_off(const void* p) {
    // generic LDS pointer: low 32 bits are the wave-relative LDS byte offset
    return (uint32_t)(uintptr_t)p;
}
#define ASYNC_CP16(lds, g, OFFS)                                               \
    asm volatile("global_load_async_to_lds_b128 %0, %1, off offset:" OFFS      \
                 :: "v"(lds), "v"(g) : "memory")
__device__ __forceinline__ void wait_async0() {
    asm volatile("s_wait_asynccnt 0x0" ::: "memory");
}

// ---------------------------------------------------------------------------
// bf16 WMMA GEMM: Out[M,N] = A[M,K] * Bt[N,K]^T (+ epilogue)
//   A, Bt bf16; Bt stored [N][K] so each lane's B-fragment is contiguous.
//   grid: (N/BN, M/BM, batches); block: 256 threads = 8 waves (wave32)
//   waves tiled 4(M) x 2(N), each wave owns 64x64 = 4x4 WMMA accumulators
//   -> 16 ds_load_b128 : 16 v_wmma per K step.
// ---------------------------------------------------------------------------
template<int EPI>
__global__ __launch_bounds__(256)
void gemm_wmma(const bf16* __restrict__ A, size_t lda, size_t sAb,
               const bf16* __restrict__ Bt, size_t ldb, size_t sBb,
               const float* __restrict__ bias,
               void* __restrict__ Ov, size_t ldo, size_t sOb,
               const bf16* __restrict__ gate, size_t ldg, size_t sGb,
               int K)
{
    __shared__ __align__(16) bf16 shA[2][BM * LDSS];
    __shared__ __align__(16) bf16 shB[2][BN * LDSS];

    const int tid = threadIdx.x;
    const int m0 = blockIdx.y * BM;
    const int n0 = blockIdx.x * BN;
    const size_t zb = blockIdx.z;

    const bf16* Ab = A + zb * sAb;
    const bf16* Bb = Bt + zb * sBb;
    const bf16* Gb = gate ? (gate + zb * sGb) : nullptr;

    // async tile loaders: A = 256 rows x 64B (1 row/thread, 4x b128)
    //                     B = 128 rows x 64B (half row/thread, 2x b128)
    const int br = tid >> 1;
    const int bh = tid & 1;

    auto issue_async = [&](int kt, int buf) {
        const size_t kk = (size_t)kt * BK;
        {
            const bf16* g = Ab + (size_t)(m0 + tid) * lda + kk;
            const uint32_t l = lds_off(&shA[buf][tid * LDSS]);
            ASYNC_CP16(l, g, "0");
            ASYNC_CP16(l, g, "16");
            ASYNC_CP16(l, g, "32");
            ASYNC_CP16(l, g, "48");
        }
        {
            const bf16* g = Bb + (size_t)(n0 + br) * ldb + kk + bh * 16;
            const uint32_t l = lds_off(&shB[buf][br * LDSS + bh * 16]);
            ASYNC_CP16(l, g, "0");
            ASYNC_CP16(l, g, "16");
        }
    };

    // wave tiling: 4x2 waves, each wave owns 64(M) x 64(N)
    const int wave = tid >> 5;
    const int lane = tid & 31;
    const int wr = (wave & 3) * 64;
    const int wc = (wave >> 2) * 64;
    const int l15 = lane & 15;
    const int lhf = lane >> 4;

    fx8 acc[4][4] = {};

    const int KT = K / BK;
    issue_async(0, 0);
    wait_async0();
    __syncthreads();

    for (int kt = 0; kt < KT; ++kt) {
        const int cur = kt & 1;
        if (kt + 1 < KT) issue_async(kt + 1, cur ^ 1);  // overlap with WMMA

        const bf16* As = shA[cur];
        const bf16* Bs = shB[cur];
        bfx16 af[4], bfr[4];
#pragma unroll
        for (int mi = 0; mi < 4; ++mi) {
            // 16-bit A layout: lane<16 -> K{0..7,16..23}; lane>=16 -> K{8..15,24..31}
            const bf16* p = As + (wr + mi * 16 + l15) * LDSS + lhf * 8;
            bfx8 lo = *(const bfx8*)p;
            bfx8 hi = *(const bfx8*)(p + 16);
            af[mi] = __builtin_shufflevector(lo, hi, 0,1,2,3,4,5,6,7,8,9,10,11,12,13,14,15);
        }
#pragma unroll
        for (int ni = 0; ni < 4; ++ni) {
            // 16-bit B layout: lane<16 -> K{0..7,16..23} of col n; lane>=16 -> K{8..15,24..31}
            const bf16* p = Bs + (wc + ni * 16 + l15) * LDSS + lhf * 8;
            bfx8 lo = *(const bfx8*)p;
            bfx8 hi = *(const bfx8*)(p + 16);
            bfr[ni] = __builtin_shufflevector(lo, hi, 0,1,2,3,4,5,6,7,8,9,10,11,12,13,14,15);
        }
#pragma unroll
        for (int mi = 0; mi < 4; ++mi)
#pragma unroll
            for (int ni = 0; ni < 4; ++ni)
                acc[mi][ni] = __builtin_amdgcn_wmma_f32_16x16x32_bf16(
                    false, af[mi], false, bfr[ni], (short)0, acc[mi][ni], false, false);

        if (kt + 1 < KT) {
            wait_async0();      // tile kt+1 resident in LDS (this wave's copies)
            __syncthreads();    // all waves' copies visible; safe to overwrite cur next iter
        }
    }

    // epilogue: C/D layout -> element r of acc is (M = r + 8*(lane>=16), N = lane&15)
    bf16*  Obf = (bf16*)Ov  + zb * sOb;
    float* Of  = (float*)Ov + zb * sOb;
#pragma unroll
    for (int mi = 0; mi < 4; ++mi) {
#pragma unroll
        for (int ni = 0; ni < 4; ++ni) {
            const int col = n0 + wc + ni * 16 + l15;
            float bv = 0.f;
            if (EPI == EPI_BF16_BIAS || EPI == EPI_BF16_BIAS_SIG ||
                EPI == EPI_BF16_BIAS_VT || EPI == EPI_F32_BIAS)
                bv = bias[col];
#pragma unroll
            for (int r = 0; r < 8; ++r) {
                const int row = m0 + wr + mi * 16 + lhf * 8 + r;
                float x = acc[mi][ni][r] + bv;
                if (EPI == EPI_BF16_BIAS) {
                    Obf[(size_t)row * ldo + col] = (bf16)x;
                } else if (EPI == EPI_BF16_BIAS_SIG) {
                    Obf[(size_t)row * ldo + col] = (bf16)(1.f / (1.f + __expf(-x)));
                } else if (EPI == EPI_BF16_BIAS_VT) {
                    // store V transposed: vT[b][d=col][s], b from global row
                    const int bidx = row >> 11;           // row / SEQ
                    const int s    = row & (SEQ - 1);
                    ((bf16*)Ov)[((size_t)bidx * DIM + col) * SEQ + s] = (bf16)x;
                } else if (EPI == EPI_F32_RAW) {
                    Of[(size_t)row * ldo + col] = x;
                } else if (EPI == EPI_BF16_GATE) {
                    float g = (float)Gb[(size_t)row * ldg + col];
                    Obf[(size_t)row * ldo + col] = (bf16)(x * g);
                } else { // EPI_F32_BIAS
                    Of[(size_t)row * ldo + col] = x;
                }
            }
        }
    }
}

// ---------------------------------------------------------------------------
// elementwise f32 -> bf16 (8 elements / thread)
// ---------------------------------------------------------------------------
__global__ __launch_bounds__(256)
void cvt_bf16(const float* __restrict__ x, bf16* __restrict__ y)
{
    const size_t i = ((size_t)blockIdx.x * 256 + threadIdx.x) * 8;
    float4 a = ((const float4*)(x + i))[0];
    float4 b = ((const float4*)(x + i))[1];
    bfx8 o = { (bf16)a.x,(bf16)a.y,(bf16)a.z,(bf16)a.w,
               (bf16)b.x,(bf16)b.y,(bf16)b.z,(bf16)b.w };
    *(bfx8*)(y + i) = o;
}

// ---------------------------------------------------------------------------
// Weight convert + transpose: W[k][n] f32 -> Wt[n][k] bf16 (32x32 tiles)
// ---------------------------------------------------------------------------
__global__ __launch_bounds__(256)
void wcvt_transpose(const float* __restrict__ W, bf16* __restrict__ Wt)
{
    __shared__ float tile[32][33];
    const int k0 = blockIdx.x * 32;
    const int n0 = blockIdx.y * 32;
    const int tx = threadIdx.x;   // 0..31
    const int ty = threadIdx.y;   // 0..7
#pragma unroll
    for (int i = 0; i < 4; ++i)
        tile[ty + i * 8][tx] = W[(size_t)(k0 + ty + i * 8) * DIM + n0 + tx];
    __syncthreads();
#pragma unroll
    for (int i = 0; i < 4; ++i)
        Wt[(size_t)(n0 + ty + i * 8) * DIM + k0 + tx] = (bf16)tile[tx][ty + i * 8];
}

// ---------------------------------------------------------------------------
// Row softmax: scores f32 [rows, SEQ] -> attn bf16, one 256-thread block/row
// ---------------------------------------------------------------------------
__global__ __launch_bounds__(256)
void softmax_rows(const float* __restrict__ scores, bf16* __restrict__ attn)
{
    const int row = blockIdx.x;
    const float* src = scores + (size_t)row * SEQ;
    bf16* dst = attn + (size_t)row * SEQ;
    const int tid = threadIdx.x;
    const int wv = tid >> 5, ln = tid & 31;

    float4 a = ((const float4*)src)[tid * 2];
    float4 b = ((const float4*)src)[tid * 2 + 1];
    float v[8] = { a.x, a.y, a.z, a.w, b.x, b.y, b.z, b.w };

    float mx = v[0];
#pragma unroll
    for (int i = 1; i < 8; ++i) mx = fmaxf(mx, v[i]);
#pragma unroll
    for (int off = 16; off; off >>= 1) mx = fmaxf(mx, __shfl_xor(mx, off, 32));

    __shared__ float red[8];
    if (ln == 0) red[wv] = mx;
    __syncthreads();
    if (tid == 0) {
        float m = red[0];
        for (int i = 1; i < 8; ++i) m = fmaxf(m, red[i]);
        red[0] = m;
    }
    __syncthreads();
    mx = red[0];
    __syncthreads();

    float sum = 0.f;
#pragma unroll
    for (int i = 0; i < 8; ++i) { v[i] = __expf(v[i] - mx); sum += v[i]; }
#pragma unroll
    for (int off = 16; off; off >>= 1) sum += __shfl_xor(sum, off, 32);
    if (ln == 0) red[wv] = sum;
    __syncthreads();
    if (tid == 0) {
        float s = 0.f;
        for (int i = 0; i < 8; ++i) s += red[i];
        red[0] = s;
    }
    __syncthreads();
    const float inv = 1.f / red[0];
#pragma unroll
    for (int i = 0; i < 8; ++i) dst[tid * 8 + i] = (bf16)(v[i] * inv);
}

// ---------------------------------------------------------------------------
extern "C" void kernel_launch(void* const* d_in, const int* in_sizes, int n_in,
                              void* d_out, int out_size, void* d_ws, size_t ws_size,
                              hipStream_t stream)
{
    const float* queries = (const float*)d_in[0];
    const float* keys    = (const float*)d_in[1];
    const float* values  = (const float*)d_in[2];
    const float* Wq = (const float*)d_in[3];  const float* bq = (const float*)d_in[4];
    const float* Wk = (const float*)d_in[5];  const float* bk = (const float*)d_in[6];
    const float* Wv = (const float*)d_in[7];  const float* bv = (const float*)d_in[8];
    const float* Wg = (const float*)d_in[9];  const float* bg = (const float*)d_in[10];
    const float* Wo = (const float*)d_in[11]; const float* bo = (const float*)d_in[12];

    const size_t BS = (size_t)NB * SEQ;   // 16384 total rows

    char* p = (char*)d_ws;
    auto alloc = [&](size_t bytes) -> char* {
        char* r = p; p += (bytes + 255) & ~(size_t)255; return r;
    };
    bf16*  xq     = (bf16*)alloc(BS * DIM * 2);           // bf16 copies of inputs
    bf16*  xk     = (bf16*)alloc(BS * DIM * 2);
    bf16*  xv     = (bf16*)alloc(BS * DIM * 2);
    bf16*  wqT    = (bf16*)alloc((size_t)DIM * DIM * 2);
    bf16*  wkT    = (bf16*)alloc((size_t)DIM * DIM * 2);
    bf16*  wvT    = (bf16*)alloc((size_t)DIM * DIM * 2);
    bf16*  wgT    = (bf16*)alloc((size_t)DIM * DIM * 2);
    bf16*  woT    = (bf16*)alloc((size_t)DIM * DIM * 2);
    bf16*  qh     = (bf16*)alloc(BS * DIM * 2);
    bf16*  kh     = (bf16*)alloc(BS * DIM * 2);
    bf16*  vT     = (bf16*)alloc(BS * DIM * 2);           // [NB][DIM][SEQ]
    bf16*  gh     = (bf16*)alloc(BS * DIM * 2);
    float* scores = (float*)alloc((size_t)NB * SEQ * SEQ * 4);
    bf16*  attn   = (bf16*)alloc((size_t)NB * SEQ * SEQ * 2);
    bf16*  hid    = (bf16*)alloc(BS * DIM * 2);

    dim3 blk(256);

    // input conversions (f32 -> bf16)
    const int cblocks = (int)((BS * DIM) / (256 * 8));    // 8192
    cvt_bf16<<<dim3(cblocks), blk, 0, stream>>>(queries, xq);
    cvt_bf16<<<dim3(cblocks), blk, 0, stream>>>(keys,    xk);
    cvt_bf16<<<dim3(cblocks), blk, 0, stream>>>(values,  xv);

    dim3 wblk(32, 8), wgrid(DIM / 32, DIM / 32);
    wcvt_transpose<<<wgrid, wblk, 0, stream>>>(Wq, wqT);
    wcvt_transpose<<<wgrid, wblk, 0, stream>>>(Wk, wkT);
    wcvt_transpose<<<wgrid, wblk, 0, stream>>>(Wv, wvT);
    wcvt_transpose<<<wgrid, wblk, 0, stream>>>(Wg, wgT);
    wcvt_transpose<<<wgrid, wblk, 0, stream>>>(Wo, woT);

    dim3 gproj(DIM / BN, BS / BM, 1);          // (8,64)

    // projections: q, k, v(transposed store), gate(sigmoid)
    gemm_wmma<EPI_BF16_BIAS><<<gproj, blk, 0, stream>>>(
        xq, DIM, 0, wqT, DIM, 0, bq, qh, DIM, 0, nullptr, 0, 0, DIM);
    gemm_wmma<EPI_BF16_BIAS><<<gproj, blk, 0, stream>>>(
        xk, DIM, 0, wkT, DIM, 0, bk, kh, DIM, 0, nullptr, 0, 0, DIM);
    gemm_wmma<EPI_BF16_BIAS_VT><<<gproj, blk, 0, stream>>>(
        xv, DIM, 0, wvT, DIM, 0, bv, vT, 0, 0, nullptr, 0, 0, DIM);
    gemm_wmma<EPI_BF16_BIAS_SIG><<<gproj, blk, 0, stream>>>(
        xq, DIM, 0, wgT, DIM, 0, bg, gh, DIM, 0, nullptr, 0, 0, DIM);

    // scores[b] = qh[b] @ kh[b]^T  (kh natural [key][d] layout == Bt [N][K])
    dim3 gsc(SEQ / BN, SEQ / BM, NB);          // (16,8,8)
    gemm_wmma<EPI_F32_RAW><<<gsc, blk, 0, stream>>>(
        qh, DIM, (size_t)SEQ * DIM, kh, DIM, (size_t)SEQ * DIM, nullptr,
        scores, SEQ, (size_t)SEQ * SEQ, nullptr, 0, 0, DIM);

    softmax_rows<<<dim3(NB * SEQ), blk, 0, stream>>>(scores, attn);

    // hid[b] = (attn[b] @ v[b]) * gate  (B operand = vT [d][s])
    dim3 gcx(DIM / BN, SEQ / BM, NB);          // (8,8,8)
    gemm_wmma<EPI_BF16_GATE><<<gcx, blk, 0, stream>>>(
        attn, SEQ, (size_t)SEQ * SEQ, vT, SEQ, (size_t)DIM * SEQ, nullptr,
        hid, DIM, (size_t)SEQ * DIM, gh, DIM, (size_t)SEQ * DIM, SEQ);

    // out = hid @ Wo + bo  (f32 output)
    gemm_wmma<EPI_F32_BIAS><<<gproj, blk, 0, stream>>>(
        hid, DIM, 0, woT, DIM, 0, bo, (float*)d_out, DIM, 0, nullptr, 0, 0, DIM);
}
